// MultiHeadAttention_15994458210437
// MI455X (gfx1250) — compile-verified, matrix operand order checked by reading, not run on hardware
//
#include <hip/hip_runtime.h>
#include <hip/hip_bf16.h>

typedef __bf16 bf16_t;
typedef __bf16 v8bf  __attribute__((ext_vector_type(8)));
typedef __bf16 v16bf __attribute__((ext_vector_type(16)));
typedef float  v8f   __attribute__((ext_vector_type(8)));

#define B_DIM 4
#define S_LEN 2048
#define D_DIM 1024
#define H_NUM 16
#define HD    64

// ---------- helpers ----------

__device__ __forceinline__ unsigned short f2bf(float f) {
  unsigned int u = __float_as_uint(f);
  u += 0x7FFFu + ((u >> 16) & 1u);     // round-to-nearest-even
  return (unsigned short)(u >> 16);
}

__device__ __forceinline__ v8f zero8() {
  v8f z = {0.0f, 0.0f, 0.0f, 0.0f, 0.0f, 0.0f, 0.0f, 0.0f};
  return z;
}

// 16x32 bf16 operand fragment per CDNA5 ISA layout:
// lane holds row (A) / column-of-B-as-row-of-Bt (B); element e maps to
// K = (e<8 ? e : e+8) + 8*half  ->  two contiguous 16B chunks at +8*half, +16+8*half.
__device__ __forceinline__ v16bf load_frag(const bf16_t* rowk, int half) {
  v8bf lo = *(const v8bf*)(rowk + 8 * half);
  v8bf hi = *(const v8bf*)(rowk + 16 + 8 * half);
  return __builtin_shufflevector(lo, hi, 0, 1, 2, 3, 4, 5, 6, 7,
                                         8, 9, 10, 11, 12, 13, 14, 15);
}

__device__ __forceinline__ v8f wmma_bf16(v16bf a, v16bf b, v8f c) {
  return __builtin_amdgcn_wmma_f32_16x16x32_bf16(false, a, false, b,
                                                 (short)0, c, false, false);
}

// Async global -> LDS copy (CDNA5, ASYNCcnt-tracked). 16 bytes per lane.
__device__ __forceinline__ void async_copy_b128(unsigned lds_off, const void* gptr) {
  asm volatile("global_load_async_to_lds_b128 %0, %1, off"
               :: "v"(lds_off), "v"(gptr)
               : "memory");
}
__device__ __forceinline__ void wait_async_le2() {
  asm volatile("s_wait_asynccnt 0x2" ::: "memory");
}
__device__ __forceinline__ void wait_async_0() {
  asm volatile("s_wait_asynccnt 0x0" ::: "memory");
}

// ---------- phase 0: conversions ----------

__global__ __launch_bounds__(256) void cvt_f32_bf16(const float* __restrict__ src,
                                                    bf16_t* __restrict__ dst, int n) {
  int i = blockIdx.x * 256 + threadIdx.x;
  if (i < n) *(unsigned short*)(dst + i) = f2bf(src[i]);
}

// Wt[n][k] = W[k][n]  (bf16), so B-operand loads are contiguous along K.
__global__ __launch_bounds__(256) void cvt_transpose(const float* __restrict__ W,
                                                     bf16_t* __restrict__ Wt) {
  int i = blockIdx.x * 256 + threadIdx.x;     // 0 .. 1M-1
  int k = i >> 10;
  int n = i & (D_DIM - 1);
  *(unsigned short*)(Wt + (size_t)n * D_DIM + k) = f2bf(W[i]);
}

// ---------- phase 1: fused QKV projection ----------
// grid (128, 16, 3), block 128 (4 waves). The block shares a 64x32 weight
// panel per K step, staged via double-buffered async global->LDS copies.
// Wave w computes the 16x64 output tile at m0 = (blockIdx.x*4+w)*16.
__global__ __launch_bounds__(128) void proj_kernel(
    const bf16_t* __restrict__ xb,
    const bf16_t* __restrict__ wtq, const bf16_t* __restrict__ wtk,
    const bf16_t* __restrict__ wtv,
    bf16_t* __restrict__ Qb, bf16_t* __restrict__ Kb, bf16_t* __restrict__ Vt) {
  __shared__ __align__(16) unsigned short bsh[2][64 * 32];   // 2 x 4KB B panels

  const int tid  = threadIdx.x;          // 0..127
  const int lane = tid & 31;
  const int wave = tid >> 5;
  const int n16 = lane & 15;
  const int half = lane >> 4;
  const int m0 = (blockIdx.x * 4 + wave) * 16;
  const int n0 = blockIdx.y * 64;
  const int z = blockIdx.z;
  const bf16_t* W = (z == 0) ? wtq : (z == 1) ? wtk : wtv;

  // per-thread async copy slots: 2 chunks of 8 bf16 (16B) each
  const int L0 = tid * 8;                // elements [L0, L0+8) of the 2048-elem panel
  const int r0 = L0 >> 5, c0 = L0 & 31;
  const int L1 = 1024 + tid * 8;
  const int r1 = L1 >> 5, c1 = L1 & 31;

  // prologue: stage panel for k0 = 0 into buffer 0
  async_copy_b128((unsigned)(uintptr_t)&bsh[0][L0], W + (size_t)(n0 + r0) * D_DIM + c0);
  async_copy_b128((unsigned)(uintptr_t)&bsh[0][L1], W + (size_t)(n0 + r1) * D_DIM + c1);

  v8f acc[4];
#pragma unroll
  for (int t = 0; t < 4; ++t) acc[t] = zero8();

  const bf16_t* arow = xb + (size_t)(m0 + n16) * D_DIM;
  for (int k0 = 0; k0 < D_DIM; k0 += 32) {
    const int cur = (k0 >> 5) & 1;
    if (k0 + 32 < D_DIM) {
      const int nxt = cur ^ 1;
      const int kn = k0 + 32;
      async_copy_b128((unsigned)(uintptr_t)&bsh[nxt][L0],
                      W + (size_t)(n0 + r0) * D_DIM + (kn + c0));
      async_copy_b128((unsigned)(uintptr_t)&bsh[nxt][L1],
                      W + (size_t)(n0 + r1) * D_DIM + (kn + c1));
      wait_async_le2();                  // oldest 2 (current panel) complete
    } else {
      wait_async_0();
    }
    __syncthreads();

    v16bf a = load_frag(arow + k0, half);
#pragma unroll
    for (int t = 0; t < 4; ++t) {
      v16bf bb = load_frag((const bf16_t*)&bsh[cur][(t * 16 + n16) * 32], half);
      acc[t] = wmma_bf16(a, bb, acc[t]);
    }
    __syncthreads();                     // panel fully consumed before overwrite
  }

#pragma unroll
  for (int t = 0; t < 4; ++t) {
    const int n = n0 + t * 16 + n16;
    const int h = n >> 6;
    const int d = n & 63;
#pragma unroll
    for (int r = 0; r < 8; ++r) {
      const int m = m0 + r + 8 * half;              // global token
      const int bi = m >> 11;                       // batch
      const int s = m & (S_LEN - 1);
      if (z == 2) {
        // V stored transposed: [b][h][d][s]
        size_t off = ((size_t)(bi * H_NUM + h) * HD + d) * S_LEN + s;
        *(unsigned short*)(Vt + off) = f2bf(acc[t][r]);
      } else {
        size_t off = ((size_t)(bi * H_NUM + h) * S_LEN + s) * HD + d;
        bf16_t* dst = (z == 0) ? Qb : Kb;
        *(unsigned short*)(dst + off) = f2bf(acc[t][r]);
      }
    }
  }
}

// ---------- phase 2: causal flash attention ----------
// grid (16, 64), block 256 (8 waves). Wave w of block x handles q-tile
// qt = x*8+w (16 query rows); blockIdx.y = b*H + h.
__global__ __launch_bounds__(256) void attn_kernel(
    const bf16_t* __restrict__ Qb, const bf16_t* __restrict__ Kb,
    const bf16_t* __restrict__ Vt, bf16_t* __restrict__ ctx) {
  __shared__ __align__(16) unsigned short pbuf[8][16 * 32];
  const int lane = threadIdx.x & 31;
  const int wave = threadIdx.x >> 5;
  const int n16 = lane & 15;
  const int half = lane >> 4;
  const int qt = blockIdx.x * 8 + wave;
  const int bh = blockIdx.y;
  const int q0 = qt * 16;

  const bf16_t* Qh = Qb + (size_t)bh * S_LEN * HD;
  const bf16_t* Kh = Kb + (size_t)bh * S_LEN * HD;
  const bf16_t* Vh = Vt + (size_t)bh * HD * S_LEN;

  // Q tile stays in registers for the whole key loop (16x64 = 2 depth frags)
  const v16bf qf0 = load_frag(Qh + (size_t)(q0 + n16) * HD + 0, half);
  const v16bf qf1 = load_frag(Qh + (size_t)(q0 + n16) * HD + 32, half);

  v8f acc[4];
#pragma unroll
  for (int t = 0; t < 4; ++t) acc[t] = zero8();
  float mrun[8], lrun[8];
#pragma unroll
  for (int r = 0; r < 8; ++r) {
    mrun[r] = -__builtin_inff();
    lrun[r] = 0.0f;
  }

  const int nchunks = (q0 + 15) / 32 + 1;      // causal bound
  for (int c = 0; c < nchunks; ++c) {
    const int kb = c * 32;
    // ---- S = Q K^T for 32 keys (two 16-key N tiles, depth 64 = 2 WMMAs each)
    v8f s0 = zero8(), s1 = zero8();
    const bf16_t* kr0 = Kh + (size_t)(kb + n16) * HD;
    const bf16_t* kr1 = Kh + (size_t)(kb + 16 + n16) * HD;
    s0 = wmma_bf16(qf0, load_frag(kr0, half), s0);
    s1 = wmma_bf16(qf0, load_frag(kr1, half), s1);
    s0 = wmma_bf16(qf1, load_frag(kr0 + 32, half), s0);
    s1 = wmma_bf16(qf1, load_frag(kr1 + 32, half), s1);

    // ---- online softmax in C-layout (row = r + 8*half, col = n16)
    const int key0 = kb + n16;
    const int key1 = key0 + 16;
    float alpha[8];
#pragma unroll
    for (int r = 0; r < 8; ++r) {
      const int qrow = q0 + r + 8 * half;
      float v0 = (key0 <= qrow) ? s0[r] * 0.125f : -__builtin_inff();
      float v1 = (key1 <= qrow) ? s1[r] * 0.125f : -__builtin_inff();
      float mx = fmaxf(v0, v1);
      mx = fmaxf(mx, __shfl_xor(mx, 1, 16));
      mx = fmaxf(mx, __shfl_xor(mx, 2, 16));
      mx = fmaxf(mx, __shfl_xor(mx, 4, 16));
      mx = fmaxf(mx, __shfl_xor(mx, 8, 16));
      const float mnew = fmaxf(mrun[r], mx);     // finite: chunk 0 always has key<=q
      const float p0 = __expf(v0 - mnew);
      const float p1 = __expf(v1 - mnew);
      alpha[r] = __expf(mrun[r] - mnew);
      mrun[r] = mnew;
      float rs = p0 + p1;
      rs += __shfl_xor(rs, 1, 16);
      rs += __shfl_xor(rs, 2, 16);
      rs += __shfl_xor(rs, 4, 16);
      rs += __shfl_xor(rs, 8, 16);
      lrun[r] = lrun[r] * alpha[r] + rs;
      // stage P (bf16) through LDS to re-layout C-frag -> A-frag
      const int m = r + 8 * half;
      pbuf[wave][m * 32 + n16] = f2bf(p0);
      pbuf[wave][m * 32 + 16 + n16] = f2bf(p1);
#pragma unroll
      for (int t = 0; t < 4; ++t) acc[t][r] *= alpha[r];
    }

    // ---- ctx += P(16x32) @ V(32x64): A-frag of P from LDS, B from Vt rows
    v16bf pf = load_frag((const bf16_t*)&pbuf[wave][n16 * 32], half);
#pragma unroll
    for (int t = 0; t < 4; ++t) {
      v16bf vf = load_frag(Vh + (size_t)(t * 16 + n16) * S_LEN + kb, half);
      acc[t] = wmma_bf16(pf, vf, acc[t]);
    }
  }

  // ---- epilogue: normalize, store ctx as [b][s][h*64+d] bf16
  const int bb = bh >> 4;
  const int hh = bh & 15;
  bf16_t* crow = ctx + ((size_t)bb * S_LEN + q0) * D_DIM + hh * HD;
#pragma unroll
  for (int r = 0; r < 8; ++r) {
    const int m = r + 8 * half;
    const float inv = 1.0f / lrun[r];
#pragma unroll
    for (int t = 0; t < 4; ++t) {
      *(unsigned short*)(crow + (size_t)m * D_DIM + t * 16 + n16) =
          f2bf(acc[t][r] * inv);
    }
  }
}

// ---------- phase 3: output projection + bias ----------
// grid (128, 16), block 128 (4 waves), same async-LDS B-panel pipeline.
__global__ __launch_bounds__(128) void outproj_kernel(
    const bf16_t* __restrict__ ctxb, const bf16_t* __restrict__ wto,
    const float* __restrict__ bo, float* __restrict__ out) {
  __shared__ __align__(16) unsigned short bsh[2][64 * 32];

  const int tid  = threadIdx.x;
  const int lane = tid & 31;
  const int wave = tid >> 5;
  const int n16 = lane & 15;
  const int half = lane >> 4;
  const int m0 = (blockIdx.x * 4 + wave) * 16;
  const int n0 = blockIdx.y * 64;

  const int L0 = tid * 8;
  const int r0 = L0 >> 5, c0 = L0 & 31;
  const int L1 = 1024 + tid * 8;
  const int r1 = L1 >> 5, c1 = L1 & 31;

  async_copy_b128((unsigned)(uintptr_t)&bsh[0][L0], wto + (size_t)(n0 + r0) * D_DIM + c0);
  async_copy_b128((unsigned)(uintptr_t)&bsh[0][L1], wto + (size_t)(n0 + r1) * D_DIM + c1);

  v8f acc[4];
#pragma unroll
  for (int t = 0; t < 4; ++t) acc[t] = zero8();

  const bf16_t* arow = ctxb + (size_t)(m0 + n16) * D_DIM;
  for (int k0 = 0; k0 < D_DIM; k0 += 32) {
    const int cur = (k0 >> 5) & 1;
    if (k0 + 32 < D_DIM) {
      const int nxt = cur ^ 1;
      const int kn = k0 + 32;
      async_copy_b128((unsigned)(uintptr_t)&bsh[nxt][L0],
                      wto + (size_t)(n0 + r0) * D_DIM + (kn + c0));
      async_copy_b128((unsigned)(uintptr_t)&bsh[nxt][L1],
                      wto + (size_t)(n0 + r1) * D_DIM + (kn + c1));
      wait_async_le2();
    } else {
      wait_async_0();
    }
    __syncthreads();

    v16bf a = load_frag(arow + k0, half);
#pragma unroll
    for (int t = 0; t < 4; ++t) {
      v16bf bb = load_frag((const bf16_t*)&bsh[cur][(t * 16 + n16) * 32], half);
      acc[t] = wmma_bf16(a, bb, acc[t]);
    }
    __syncthreads();
  }

#pragma unroll
  for (int t = 0; t < 4; ++t) {
    const int n = n0 + t * 16 + n16;
    const float bias = bo[n];
#pragma unroll
    for (int r = 0; r < 8; ++r) {
      const int m = m0 + r + 8 * half;
      out[(size_t)m * D_DIM + n] = acc[t][r] + bias;
    }
  }
}

// ---------- launch ----------

extern "C" void kernel_launch(void* const* d_in, const int* in_sizes, int n_in,
                              void* d_out, int out_size, void* d_ws, size_t ws_size,
                              hipStream_t stream) {
  (void)in_sizes; (void)n_in; (void)out_size; (void)ws_size;
  const float* x  = (const float*)d_in[0];
  const float* Wq = (const float*)d_in[1];
  const float* Wk = (const float*)d_in[2];
  const float* Wv = (const float*)d_in[3];
  const float* Wo = (const float*)d_in[4];
  const float* bo = (const float*)d_in[5];
  float* out = (float*)d_out;

  const size_t MD = (size_t)B_DIM * S_LEN * D_DIM;  // 8,388,608
  const size_t WD = (size_t)D_DIM * D_DIM;          // 1,048,576

  char* p = (char*)d_ws;
  bf16_t* xb   = (bf16_t*)p; p += MD * 2;
  bf16_t* wtq  = (bf16_t*)p; p += WD * 2;
  bf16_t* wtk  = (bf16_t*)p; p += WD * 2;
  bf16_t* wtv  = (bf16_t*)p; p += WD * 2;
  bf16_t* wto  = (bf16_t*)p; p += WD * 2;
  bf16_t* Qb   = (bf16_t*)p; p += MD * 2;
  bf16_t* Kb   = (bf16_t*)p; p += MD * 2;
  bf16_t* Vt   = (bf16_t*)p; p += MD * 2;
  bf16_t* ctxb = (bf16_t*)p; p += MD * 2;

  cvt_f32_bf16<<<(int)(MD / 256), 256, 0, stream>>>(x, xb, (int)MD);
  cvt_transpose<<<(int)(WD / 256), 256, 0, stream>>>(Wq, wtq);
  cvt_transpose<<<(int)(WD / 256), 256, 0, stream>>>(Wk, wtk);
  cvt_transpose<<<(int)(WD / 256), 256, 0, stream>>>(Wv, wtv);
  cvt_transpose<<<(int)(WD / 256), 256, 0, stream>>>(Wo, wto);

  proj_kernel<<<dim3(128, 16, 3), 128, 0, stream>>>(xb, wtq, wtk, wtv, Qb, Kb, Vt);
  attn_kernel<<<dim3(16, 64), 256, 0, stream>>>(Qb, Kb, Vt, ctxb);
  outproj_kernel<<<dim3(128, 16), 128, 0, stream>>>(ctxb, wto, bo, out);
}
